// LSTM_11922829214535
// MI455X (gfx1250) — matrix operand order repaired and checked by reading, not compiled, not run on hardware
//
#include <hip/hip_runtime.h>
#include <hip/hip_bf16.h>
#include <math.h>

// ---------------------------------------------------------------------------
// LSTM (T=4096, D=H=2048, C=10) for MI455X / gfx1250, wave32.
//
// Phase 1: Xproj[T,4H] = X @ Wx^T + b  -> TDM (tensor_load_to_lds) double-
//          buffered LDS tiles + WMMA bf16 (137 GFLOP), s_wait_tensorcnt sync.
// Phase 2: sequential recurrence -> persistent kernel; Wh (32MB bf16) lives in
//          the 192MB L2 and is re-streamed each of 4096 steps (131 GB from L2,
//          ~0.4 GB from HBM).
// Phase 3: classify last output gate, log-softmax (tiny).
// ---------------------------------------------------------------------------

typedef __attribute__((ext_vector_type(16))) __bf16 v16bf;
typedef __attribute__((ext_vector_type(8)))  __bf16 v8bf;
typedef __attribute__((ext_vector_type(8)))  float  v8f;
typedef __attribute__((ext_vector_type(4)))  unsigned int u32x4;
typedef __attribute__((ext_vector_type(8)))  int i32x8;
typedef __attribute__((ext_vector_type(4)))  int i32x4;

#define T_SEQ 4096
#define DDIM  2048
#define HDIM  2048
#define G4H   8192   // 4*H
#define DH    4096   // D+H
#define NCLS  10

// Workspace layout (bytes). Total ~235 MB required.
#define OFF_WXB   (0UL)                         // 8192*2048 bf16 = 32 MB
#define OFF_WHB   (33554432UL)                  // 8192*2048 bf16 = 32 MB
#define OFF_XB    (67108864UL)                  // 4096*2048 bf16 = 16 MB
#define OFF_BFU   (83886080UL)                  // 8192 f32
#define OFF_HBUF  (83918848UL)                  // 2*2048 f32 (double buffer)
#define OFF_OLAST (83935232UL)                  // 2048 f32
#define OFF_BAR   (83943424UL)                  // 2 u32 barrier state
#define OFF_XPROJ (100663296UL)                 // 4096*8192 f32 = 128 MB

__device__ __forceinline__ unsigned short f2bf_rne(float f) {
    unsigned int u = __float_as_uint(f);
    u += 0x7FFFu + ((u >> 16) & 1u);           // round-to-nearest-even
    return (unsigned short)(u >> 16);
}

// ---------------------------------------------------------------------------
// TDM: 2D tile load Global->LDS. D# packed per CDNA5 ISA.
// data_size = 2 bytes; LDS padding: every 16 dwords (64B) insert 4 dwords
// (16B)  => 64B tile rows land with an 80B stride in LDS (bank-friendly,
// 16B aligned for ds_load_b128).
// ---------------------------------------------------------------------------
__device__ __forceinline__ void tdm_load_tile_2d(unsigned lds_addr, unsigned long long gaddr,
                                                 unsigned tile_d0, unsigned tile_d1,
                                                 unsigned tens_d0, unsigned tens_d1,
                                                 unsigned stride0_elems) {
    u32x4 g0;
    g0[0] = 1u;                                            // count=1, user desc
    g0[1] = lds_addr;                                      // LDS byte address
    g0[2] = (unsigned)(gaddr & 0xFFFFFFFFull);             // global_addr[31:0]
    g0[3] = (unsigned)((gaddr >> 32) & 0x01FFFFFFull)      // global_addr[56:32]
          | 0x80000000u;                                   // type=2 ("image")
    i32x8 g1;
    g1[0] = (int)((1u << 16)        /* data_size: 2 bytes      */
                | (1u << 20)        /* pad_enable               */
                | (3u << 22)        /* pad_interval: 16 dwords  */
                | (3u << 25));      /* pad_amount:   4 dwords   */
    g1[1] = (int)((tens_d0 & 0xFFFFu) << 16);              // tensor_dim0[15:0]
    g1[2] = (int)((tens_d0 >> 16) | ((tens_d1 & 0xFFFFu) << 16));
    g1[3] = (int)((tens_d1 >> 16) | (tile_d0 << 16));      // tile_dim0
    g1[4] = (int)(tile_d1 & 0xFFFFu);                      // tile_dim1 (tile_dim2=0)
    g1[5] = (int)stride0_elems;                            // tensor_dim0_stride[31:0]
    g1[6] = 0;
    g1[7] = 0;
    i32x4 z4 = {0, 0, 0, 0};
#if __clang_major__ >= 23
    i32x8 z8 = {0, 0, 0, 0, 0, 0, 0, 0};
    __builtin_amdgcn_tensor_load_to_lds(g0, g1, z4, z4, z8, 0);
#else
    __builtin_amdgcn_tensor_load_to_lds(g0, g1, z4, z4, 0);
#endif
}

// ---------------------------------------------------------------------------
// Conversion kernels
// ---------------------------------------------------------------------------
__global__ __launch_bounds__(256)
void cvt_input_kernel(const float* __restrict__ X, unsigned short* __restrict__ Xb) {
    size_t idx = (size_t)blockIdx.x * 256 + threadIdx.x;
    if (idx < (size_t)T_SEQ * DDIM) Xb[idx] = f2bf_rne(X[idx]);
}

__global__ __launch_bounds__(256)
void cvt_weights_kernel(const float* __restrict__ Wf, const float* __restrict__ Wi,
                        const float* __restrict__ Wc, const float* __restrict__ Wo,
                        unsigned short* __restrict__ WxB, unsigned short* __restrict__ WhB) {
    size_t idx = (size_t)blockIdx.x * 256 + threadIdx.x;
    if (idx >= (size_t)G4H * DH) return;
    int r = (int)(idx >> 12);
    int k = (int)(idx & 4095);
    int gate = r >> 11;
    int j = r & 2047;
    const float* W = (gate == 0) ? Wf : (gate == 1) ? Wi : (gate == 2) ? Wc : Wo;
    unsigned short b = f2bf_rne(W[(size_t)j * DH + k]);
    if (k < DDIM) WxB[(size_t)r * DDIM + k] = b;
    else          WhB[(size_t)r * HDIM + (k - DDIM)] = b;
}

__global__ __launch_bounds__(256)
void init_state_kernel(float* __restrict__ hbuf, unsigned* __restrict__ bar,
                       float* __restrict__ bfu,
                       const float* __restrict__ bf, const float* __restrict__ bi,
                       const float* __restrict__ bc, const float* __restrict__ bo) {
    int tid = threadIdx.x;
    for (int i = tid; i < 2 * HDIM; i += 256) hbuf[i] = 0.0f;
    for (int i = tid; i < G4H; i += 256) {
        float v = (i < 2048) ? bf[i] : (i < 4096) ? bi[i - 2048]
                : (i < 6144) ? bc[i - 4096] : bo[i - 6144];
        bfu[i] = v;
    }
    if (tid < 2) bar[tid] = 0u;
}

// ---------------------------------------------------------------------------
// Phase 1: Xproj = X @ Wx^T + b. Block = 256 thr (8 waves), tile 64(M)x128(N).
// Waves: 4(M)x2(N), each wave 16x64 (4 wmma accumulators). K chunks of 32.
// Wave 0 drives the TDM double-buffer; everyone reads fragments from LDS.
// LDS: A bufs 2x(64 rows x 80B), B bufs 2x(128 rows x 80B) = 30 KB.
// Grid = (8192/128, 4096/64) = (64, 64).
// ---------------------------------------------------------------------------
#define A_ROWB 80
#define A_BUFB (64 * A_ROWB)       // 5120
#define B_ROWB 80
#define B_BUFB (128 * B_ROWB)      // 10240

__global__ __launch_bounds__(256)
void xproj_gemm_kernel(const __bf16* __restrict__ Xb, const __bf16* __restrict__ WxB,
                       const float* __restrict__ bfu, float* __restrict__ Xproj) {
    __shared__ __align__(16) unsigned char smem[2 * A_BUFB + 2 * B_BUFB];

    const int lane = threadIdx.x & 31;
    const int wv   = threadIdx.x >> 5;     // 0..7
    const int l16  = lane & 15;
    const int lh   = lane >> 4;            // 0/1
    const int mw   = wv >> 1;              // 0..3 (M)
    const int nw   = wv & 1;               // 0..1 (N)
    const int m_base = blockIdx.y << 6;    // 64 rows
    const int n_base = blockIdx.x << 7;    // 128 cols

    const unsigned ldsA[2] = { (unsigned)(size_t)&smem[0],
                               (unsigned)(size_t)&smem[A_BUFB] };
    const unsigned ldsB[2] = { (unsigned)(size_t)&smem[2 * A_BUFB],
                               (unsigned)(size_t)&smem[2 * A_BUFB + B_BUFB] };

    const unsigned long long gA = (unsigned long long)(Xb  + (size_t)m_base * DDIM);
    const unsigned long long gB = (unsigned long long)(WxB + (size_t)n_base * DDIM);

    // Prologue: prime both LDS buffers (chunks 0 and 1). 2 TDM ops per chunk.
    if (wv == 0) {
        tdm_load_tile_2d(ldsA[0], gA,        32u, 64u,  DDIM, T_SEQ, DDIM);
        tdm_load_tile_2d(ldsB[0], gB,        32u, 128u, DDIM, G4H,   DDIM);
        tdm_load_tile_2d(ldsA[1], gA + 64u,  32u, 64u,  DDIM, T_SEQ, DDIM);
        tdm_load_tile_2d(ldsB[1], gB + 64u,  32u, 128u, DDIM, G4H,   DDIM);
    }

    v8f acc[4] = {v8f{}, v8f{}, v8f{}, v8f{}};

    // Per-lane LDS fragment base offsets (byte offsets within a buffer).
    const int aOff = (mw * 16 + l16) * A_ROWB + lh * 16;       // + {0,32}
    const int bRow = nw * 64 + l16;                            // + nt*16 rows

    for (int c = 0; c < 64; ++c) {
        const int buf = c & 1;
        if (wv == 0) {
            // TDM ops from one wave complete in order: <=2 outstanding means
            // the pair for chunk c has landed.
            if (c < 63) __builtin_amdgcn_s_wait_tensorcnt(2);
            else        __builtin_amdgcn_s_wait_tensorcnt(0);
        }
        __syncthreads();                                       // buffer `buf` ready

        const unsigned char* sA = smem + buf * A_BUFB + aOff;
        const unsigned char* sB = smem + 2 * A_BUFB + buf * B_BUFB;

        // A fragment (16x32): lane half 0 -> K {0..7,16..23}, half 1 -> {8..15,24..31}
        v8bf alo = *(const v8bf*)(sA);
        v8bf ahi = *(const v8bf*)(sA + 32);
        v16bf a;
#pragma unroll
        for (int i = 0; i < 8; ++i) { a[i] = alo[i]; a[i + 8] = ahi[i]; }

#pragma unroll
        for (int nt = 0; nt < 4; ++nt) {
            const unsigned char* sBp = sB + (bRow + nt * 16) * B_ROWB + lh * 16;
            v8bf b0 = *(const v8bf*)(sBp);
            v8bf b1 = *(const v8bf*)(sBp + 32);
            v16bf b;
#pragma unroll
            for (int i = 0; i < 8; ++i) { b[i] = b0[i]; b[i + 8] = b1[i]; }
            acc[nt] = __builtin_amdgcn_wmma_f32_16x16x32_bf16(
                false, a, false, b, (short)0, acc[nt], false, false);
        }

        __syncthreads();                                       // done reading `buf`
        if (wv == 0 && c + 2 < 64) {
            unsigned long long koff = (unsigned long long)(c + 2) * 64ull; // 32 elems * 2B
            tdm_load_tile_2d(ldsA[buf], gA + koff, 32u, 64u,  DDIM, T_SEQ, DDIM);
            tdm_load_tile_2d(ldsB[buf], gB + koff, 32u, 128u, DDIM, G4H,   DDIM);
        }
    }

    // C/D layout: VGPR i, lanes 0-15 -> M=i, N=lane; lanes 16-31 -> M=8+i.
    const int mrow = m_base + mw * 16 + lh * 8;
#pragma unroll
    for (int nt = 0; nt < 4; ++nt) {
        int n = n_base + nw * 64 + nt * 16 + l16;
        float bv = bfu[n];
#pragma unroll
        for (int i = 0; i < 8; ++i)
            Xproj[(size_t)(mrow + i) * G4H + n] = acc[nt][i] + bv;
    }
}

// ---------------------------------------------------------------------------
// Phase 2: persistent recurrent kernel. 256 blocks x 256 threads. Block b owns
// 8 gate-quads q0 = 8*b. Per step: 32 Wh rows x 2048 cols, 8 threads per row.
// Grid-wide sense barrier between steps; h double-buffered.
// ---------------------------------------------------------------------------
__global__ __launch_bounds__(256)
void lstm_recurrent_kernel(const __bf16* __restrict__ WhB, const float* __restrict__ Xproj,
                           float* __restrict__ hbuf, float* __restrict__ olast,
                           unsigned* __restrict__ bar) {
    __shared__ float hLDS[HDIM];
    __shared__ float part[32][9];
    __shared__ float gLDS[32];

    const int tid  = threadIdx.x;
    const int q0   = blockIdx.x * 8;
    const int lr   = tid >> 3;
    const int seg  = tid & 7;
    const int gate = lr >> 3;
    const int qi   = lr & 7;
    const int row  = gate * HDIM + q0 + qi;
    const unsigned* wrow = (const unsigned*)(WhB + (size_t)row * HDIM) + seg * 128;

    unsigned* cnt   = bar;
    unsigned* phase = bar + 1;
    float creg = 0.0f;

    for (unsigned t = 0; t < T_SEQ; ++t) {
        const float* hin  = hbuf + (t & 1u) * HDIM;
        float*       hout = hbuf + ((t + 1u) & 1u) * HDIM;

        for (int i = tid; i < HDIM; i += 256) hLDS[i] = hin[i];
        __builtin_prefetch((const void*)wrow, 0, 3);   // global_prefetch_b8 (L2 hit)
        __syncthreads();

        const float* hp = &hLDS[seg * 256];
        float acc = 0.0f;
#pragma unroll 4
        for (int kk = 0; kk < 128; kk += 4) {
            uint4 w4 = *(const uint4*)(wrow + kk);
            const float* h8 = hp + kk * 2;
            acc += __uint_as_float(w4.x << 16)          * h8[0];
            acc += __uint_as_float(w4.x & 0xFFFF0000u) * h8[1];
            acc += __uint_as_float(w4.y << 16)          * h8[2];
            acc += __uint_as_float(w4.y & 0xFFFF0000u) * h8[3];
            acc += __uint_as_float(w4.z << 16)          * h8[4];
            acc += __uint_as_float(w4.z & 0xFFFF0000u) * h8[5];
            acc += __uint_as_float(w4.w << 16)          * h8[6];
            acc += __uint_as_float(w4.w & 0xFFFF0000u) * h8[7];
        }
        part[lr][seg] = acc;
        __syncthreads();

        if (tid < 32) {
            float s = Xproj[(size_t)t * G4H + (tid >> 3) * HDIM + q0 + (tid & 7)];
#pragma unroll
            for (int j = 0; j < 8; ++j) s += part[tid][j];
            gLDS[tid] = s;
        }
        __syncthreads();

        if (tid < 8) {
            float fg = 1.0f / (1.0f + __expf(-gLDS[tid]));
            float ig = 1.0f / (1.0f + __expf(-gLDS[8 + tid]));
            float cu = tanhf(gLDS[16 + tid]);
            float og = 1.0f / (1.0f + __expf(-gLDS[24 + tid]));
            creg = fg * creg + ig * cu;
            hout[q0 + tid] = og * tanhf(creg);
            if (t == T_SEQ - 1) olast[q0 + tid] = og;
        }

        __syncthreads();
        if (tid == 0) {
            __threadfence();
            if (atomicAdd(cnt, 1u) == (unsigned)(gridDim.x - 1)) {
                atomicExch(cnt, 0u);
                atomicAdd(phase, 1u);
            } else {
                while (atomicAdd(phase, 0u) < t + 1u) __builtin_amdgcn_s_sleep(2);
            }
            __threadfence();
        }
        __syncthreads();
    }
}

// ---------------------------------------------------------------------------
// Phase 3: logits = o_last @ Wcls^T + bcls ; log_softmax. One wave.
// ---------------------------------------------------------------------------
__global__ void classify_kernel(const float* __restrict__ olast,
                                const float* __restrict__ Wcls,
                                const float* __restrict__ bcls,
                                float* __restrict__ out) {
    int lane = threadIdx.x;
    float logits[NCLS];
#pragma unroll
    for (int c = 0; c < NCLS; ++c) {
        float acc = 0.0f;
        for (int j = lane; j < HDIM; j += 32)
            acc += olast[j] * Wcls[(size_t)c * HDIM + j];
        for (int off = 16; off > 0; off >>= 1)
            acc += __shfl_down(acc, off, 32);
        logits[c] = acc;
    }
    if (lane == 0) {
        float mx = -3.4e38f;
#pragma unroll
        for (int c = 0; c < NCLS; ++c) { logits[c] += bcls[c]; mx = fmaxf(mx, logits[c]); }
        float s = 0.0f;
#pragma unroll
        for (int c = 0; c < NCLS; ++c) s += __expf(logits[c] - mx);
        float lse = mx + __logf(s);
#pragma unroll
        for (int c = 0; c < NCLS; ++c) out[c] = logits[c] - lse;
    }
}

// ---------------------------------------------------------------------------
extern "C" void kernel_launch(void* const* d_in, const int* in_sizes, int n_in,
                              void* d_out, int out_size, void* d_ws, size_t ws_size,
                              hipStream_t stream) {
    const float* X    = (const float*)d_in[0];
    const float* Wf   = (const float*)d_in[1];
    const float* bf_  = (const float*)d_in[2];
    const float* Wi   = (const float*)d_in[3];
    const float* bi_  = (const float*)d_in[4];
    const float* Wc   = (const float*)d_in[5];
    const float* bc_  = (const float*)d_in[6];
    const float* Wo   = (const float*)d_in[7];
    const float* bo_  = (const float*)d_in[8];
    const float* Wcls = (const float*)d_in[9];
    const float* bcls = (const float*)d_in[10];
    float* out = (float*)d_out;

    char* ws = (char*)d_ws;
    unsigned short* WxB = (unsigned short*)(ws + OFF_WXB);
    unsigned short* WhB = (unsigned short*)(ws + OFF_WHB);
    unsigned short* Xb  = (unsigned short*)(ws + OFF_XB);
    float*    bfu   = (float*)(ws + OFF_BFU);
    float*    hbuf  = (float*)(ws + OFF_HBUF);
    float*    olast = (float*)(ws + OFF_OLAST);
    unsigned* bar   = (unsigned*)(ws + OFF_BAR);
    float*    Xproj = (float*)(ws + OFF_XPROJ);

    cvt_input_kernel<<<(T_SEQ * DDIM) / 256, 256, 0, stream>>>(X, Xb);
    cvt_weights_kernel<<<(G4H * DH) / 256, 256, 0, stream>>>(Wf, Wi, Wc, Wo, WxB, WhB);
    init_state_kernel<<<1, 256, 0, stream>>>(hbuf, bar, bfu, bf_, bi_, bc_, bo_);

    xproj_gemm_kernel<<<dim3(G4H / 128, T_SEQ / 64), 256, 0, stream>>>(
        (const __bf16*)Xb, (const __bf16*)WxB, bfu, Xproj);

    lstm_recurrent_kernel<<<256, 256, 0, stream>>>(
        (const __bf16*)WhB, Xproj, hbuf, olast, bar);

    classify_kernel<<<1, 32, 0, stream>>>(olast, Wcls, bcls, out);

    (void)in_sizes; (void)n_in; (void)out_size; (void)ws_size;
}
